// EnhancedSTGNN_36352603194076
// MI455X (gfx1250) — compile-verified, matrix operand order checked by reading, not run on hardware
//
#include <hip/hip_runtime.h>
#include <hip/hip_bf16.h>

typedef __attribute__((ext_vector_type(16))) _Float16 v16h;
typedef __attribute__((ext_vector_type(8)))  float    v8f;

#define Bd   16
#define Td   24
#define Nd   207
#define Cd   64
#define NHd  4
#define HDd  16
#define HORd 12
#define TOKd (Bd*Td*Nd)   /* 79488 */
#define BTd  (Bd*Td)      /* 384   */
#define BNd  (Bd*Nd)      /* 3312  */

// ---------------------------------------------------------------------------
// WMMA GEMM:  C[m,n] = sum_k A[m,k] * B[n,k]  (+bias[n], opt ReLU)
// A row-major (M,K) ld=K ; B row-major (N,K) ld=K ; C row-major (M,N) ld=N
// batched via blockIdx.z (element strides sA/sB/sC).
// One wave computes a 16x32 tile: one A fragment feeds two
// v_wmma_f32_16x16x32_f16 per 32-wide K chunk.
// Fragment element e of lane L maps to k = (e&7) + (L>=16 ? 8:0) + (e>=8 ? 16:0),
// i.e. two runs of 8 consecutive k -> four float4 loads per fragment when K%4==0.
// ---------------------------------------------------------------------------
template <bool RELU>
__global__ __launch_bounds__(32)
void gemm_nt_kernel(const float* __restrict__ A, const float* __restrict__ Bm,
                    const float* __restrict__ bias, float* __restrict__ C,
                    int M, int N, int K, long sA, long sB, long sC) {
  const int z = blockIdx.z;
  A  += (long)z * sA;
  Bm += (long)z * sB;
  C  += (long)z * sC;
  const int lane = threadIdx.x;           // 0..31
  const int l15  = lane & 15;
  const int hi8  = (lane >> 4) << 3;      // 0 or 8 (K sub-half per ISA layout)
  const int tm   = blockIdx.y << 4;
  const int tn   = blockIdx.x << 5;
  const int row  = tm + l15;
  const int col0 = tn + l15;
  const int col1 = tn + 16 + l15;

  const bool fullMN = (tm + 16 <= M) && (tn + 32 <= N);
  const bool vec4   = fullMN && ((K & 3) == 0);

  const float* __restrict__ Arow = A  + (long)(row  < M ? row  : M - 1) * K;
  const float* __restrict__ Br0  = Bm + (long)(col0 < N ? col0 : N - 1) * K;
  const float* __restrict__ Br1  = Bm + (long)(col1 < N ? col1 : N - 1) * K;

  v8f acc0 = {}, acc1 = {};
  int k0 = 0;

#define PACK8(dst, off, f0, f1)                                   \
  dst[(off) + 0] = (_Float16)(f0).x; dst[(off) + 1] = (_Float16)(f0).y; \
  dst[(off) + 2] = (_Float16)(f0).z; dst[(off) + 3] = (_Float16)(f0).w; \
  dst[(off) + 4] = (_Float16)(f1).x; dst[(off) + 5] = (_Float16)(f1).y; \
  dst[(off) + 6] = (_Float16)(f1).z; dst[(off) + 7] = (_Float16)(f1).w;

  if (vec4) {
    for (; k0 + 32 <= K; k0 += 32) {
      const int kb = k0 + hi8;
      const float4 a0 = *(const float4*)(Arow + kb);
      const float4 a1 = *(const float4*)(Arow + kb + 4);
      const float4 a2 = *(const float4*)(Arow + kb + 16);
      const float4 a3 = *(const float4*)(Arow + kb + 20);
      const float4 p0 = *(const float4*)(Br0 + kb);
      const float4 p1 = *(const float4*)(Br0 + kb + 4);
      const float4 p2 = *(const float4*)(Br0 + kb + 16);
      const float4 p3 = *(const float4*)(Br0 + kb + 20);
      const float4 q0 = *(const float4*)(Br1 + kb);
      const float4 q1 = *(const float4*)(Br1 + kb + 4);
      const float4 q2 = *(const float4*)(Br1 + kb + 16);
      const float4 q3 = *(const float4*)(Br1 + kb + 20);
      v16h a, b0, b1;
      PACK8(a, 0, a0, a1)  PACK8(a, 8, a2, a3)
      PACK8(b0, 0, p0, p1) PACK8(b0, 8, p2, p3)
      PACK8(b1, 0, q0, q1) PACK8(b1, 8, q2, q3)
      acc0 = __builtin_amdgcn_wmma_f32_16x16x32_f16(false, a, false, b0, (short)0, acc0, false, false);
      acc1 = __builtin_amdgcn_wmma_f32_16x16x32_f16(false, a, false, b1, (short)0, acc1, false, false);
    }
  } else if (fullMN) {
    // full tile, K not vectorizable (e.g. K=207): plain scalar loads, no masks
    for (; k0 + 32 <= K; k0 += 32) {
      v16h a, b0, b1;
#pragma unroll
      for (int e = 0; e < 16; ++e) {
        const int k = k0 + hi8 + (e & 7) + ((e & 8) ? 16 : 0);
        a[e]  = (_Float16)Arow[k];
        b0[e] = (_Float16)Br0[k];
        b1[e] = (_Float16)Br1[k];
      }
      acc0 = __builtin_amdgcn_wmma_f32_16x16x32_f16(false, a, false, b0, (short)0, acc0, false, false);
      acc1 = __builtin_amdgcn_wmma_f32_16x16x32_f16(false, a, false, b1, (short)0, acc1, false, false);
    }
  }
  // edge tiles and K tail: clamped-index + select-to-zero (EXEC stays full)
  for (; k0 < K; k0 += 32) {
    const float am  = (row  < M) ? 1.f : 0.f;
    const float bm0 = (col0 < N) ? 1.f : 0.f;
    const float bm1 = (col1 < N) ? 1.f : 0.f;
    v16h a, b0, b1;
#pragma unroll
    for (int e = 0; e < 16; ++e) {
      const int k  = k0 + hi8 + (e & 7) + ((e & 8) ? 16 : 0);
      const int kc = k < K ? k : K - 1;
      const float km = (k < K) ? 1.f : 0.f;
      a[e]  = (_Float16)(Arow[kc] * (am * km));
      b0[e] = (_Float16)(Br0[kc] * (bm0 * km));
      b1[e] = (_Float16)(Br1[kc] * (bm1 * km));
    }
    acc0 = __builtin_amdgcn_wmma_f32_16x16x32_f16(false, a, false, b0, (short)0, acc0, false, false);
    acc1 = __builtin_amdgcn_wmma_f32_16x16x32_f16(false, a, false, b1, (short)0, acc1, false, false);
  }
#undef PACK8

  // epilogue (C/D layout: VGPR r holds row tm + r + (lane>=16 ? 8:0), col = lane&15)
  const int rbase = tm + hi8;
  if (fullMN) {
    const float bv0 = bias ? bias[col0] : 0.f;
    const float bv1 = bias ? bias[col1] : 0.f;
#pragma unroll
    for (int r = 0; r < 8; ++r) {
      float v0 = acc0[r] + bv0;
      float v1 = acc1[r] + bv1;
      if (RELU) { v0 = fmaxf(v0, 0.f); v1 = fmaxf(v1, 0.f); }
      C[(long)(rbase + r) * N + col0] = v0;
      C[(long)(rbase + r) * N + col1] = v1;
    }
  } else {
    const float bv0 = (bias && col0 < N) ? bias[col0] : 0.f;
    const float bv1 = (bias && col1 < N) ? bias[col1] : 0.f;
#pragma unroll
    for (int r = 0; r < 8; ++r) {
      const int cr = rbase + r;
      if (cr < M) {
        if (col0 < N) {
          float v0 = acc0[r] + bv0;
          if (RELU) v0 = fmaxf(v0, 0.f);
          C[(long)cr * N + col0] = v0;
        }
        if (col1 < N) {
          float v1 = acc1[r] + bv1;
          if (RELU) v1 = fmaxf(v1, 0.f);
          C[(long)cr * N + col1] = v1;
        }
      }
    }
  }
}

static void gemm(const float* A, const float* Bm, const float* bias, float* C,
                 int M, int N, int K, int batch, long sA, long sB, long sC,
                 bool relu, hipStream_t s) {
  dim3 g((unsigned)((N + 31) / 32), (unsigned)((M + 15) / 16), (unsigned)batch);
  if (relu) gemm_nt_kernel<true ><<<g, dim3(32), 0, s>>>(A, Bm, bias, C, M, N, K, sA, sB, sC);
  else      gemm_nt_kernel<false><<<g, dim3(32), 0, s>>>(A, Bm, bias, C, M, N, K, sA, sB, sC);
}

// ---------------------------------------------------------------------------
// Block reductions (blockDim.x == 256)
// ---------------------------------------------------------------------------
__device__ __forceinline__ float blk_max256(float v, float* red, int tid) {
  red[tid] = v; __syncthreads();
  for (int o = 128; o > 0; o >>= 1) {
    if (tid < o) red[tid] = fmaxf(red[tid], red[tid + o]);
    __syncthreads();
  }
  const float r = red[0]; __syncthreads();
  return r;
}
__device__ __forceinline__ float blk_sum256(float v, float* red, int tid) {
  red[tid] = v; __syncthreads();
  for (int o = 128; o > 0; o >>= 1) {
    if (tid < o) red[tid] += red[tid + o];
    __syncthreads();
  }
  const float r = red[0]; __syncthreads();
  return r;
}

// ---------------------------------------------------------------------------
// Input embedding + positional encoding.  h token-major (B,T,N,C).
// ---------------------------------------------------------------------------
__global__ void embed_kernel(const float* __restrict__ x, const float* __restrict__ nodeE,
                             const float* __restrict__ hw, const float* __restrict__ hb,
                             const float* __restrict__ dw, const float* __restrict__ db,
                             const float* __restrict__ ww, const float* __restrict__ wb,
                             const float* __restrict__ inw, const float* __restrict__ inb,
                             float* __restrict__ h) {
  const long tok = blockIdx.x;
  const int  c   = threadIdx.x;
  const int  n   = (int)(tok % Nd);
  const long bt  = tok / Nd;
  const int  t   = (int)(bt % Td);
  const float* xr = x + tok * 4;
  const float flow = xr[0], hour = xr[1], wknd = xr[2], day = xr[3];
  const float* w = inw + c * 65;  // [flow | node(16) | hour(16) | day(16) | weekend(16)]
  float s = inb[c] + w[0] * flow;
#pragma unroll
  for (int e = 0; e < 16; ++e) {
    s += w[1 + e]  * nodeE[n * 16 + e];
    s += w[17 + e] * (hour * hw[e] + hb[e]);
    s += w[33 + e] * (day  * dw[e] + db[e]);
    s += w[49 + e] * (wknd * ww[e] + wb[e]);
  }
  const int   i2  = c & ~1;
  const float dv  = expf((float)i2 * (-9.210340371976184f / 64.f));
  const float ang = (float)t * dv;
  s += (c & 1) ? cosf(ang) : sinf(ang);
  h[tok * Cd + c] = s;
}

// adj_norm[i][j] = adj[i][j] / max(sum_j |adj[i][j]|, 1e-12)
__global__ void adjnorm_kernel(const float* __restrict__ adj, float* __restrict__ adjn) {
  const int i = blockIdx.x, tid = threadIdx.x;
  __shared__ float red[256];
  float s = (tid < Nd) ? fabsf(adj[i * Nd + tid]) : 0.f;
  const float tot = blk_sum256(s, red, tid);
  const float inv = 1.f / fmaxf(tot, 1e-12f);
  if (tid < Nd) adjn[i * Nd + tid] = adj[i * Nd + tid] * inv;
}

// ---------------------------------------------------------------------------
// Spatial attention row: scrambled batch p (reference reshape semantics),
// scores over 8 elems, softmax -> mask(-inf) -> softmax, write to d_out.
// q8/k8 are token-major (B,T,N,8).
// ---------------------------------------------------------------------------
__global__ void spatial_attn_kernel(const float* __restrict__ q8, const float* __restrict__ k8,
                                    const float* __restrict__ adjn, float* __restrict__ attn_out) {
  const int p   = blockIdx.x;   // 0..383 (scrambled (b,o,t_hi))
  const int n   = blockIdx.y;   // 0..206
  const int tid = threadIdx.x;  // 0..255 (m)
  const int b = p / Td, rem = p % Td, o = rem / 3, thi = rem % 3;
  __shared__ float red[256];
  float qv[8];
#pragma unroll
  for (int oo = 0; oo < 8; ++oo) {
    const int t = thi * 8 + oo;
    qv[oo] = q8[((long)((b * Td + t) * Nd + n)) * 8 + o];
  }
  float score = -1e30f;
  if (tid < Nd) {
    float d = 0.f;
#pragma unroll
    for (int oo = 0; oo < 8; ++oo) {
      const int t = thi * 8 + oo;
      d += qv[oo] * k8[((long)((b * Td + t) * Nd + tid)) * 8 + o];
    }
    score = d * 0.125f;  // 1/sqrt(64)
  }
  // softmax #1
  const float mx = blk_max256(score, red, tid);
  const float e1 = (tid < Nd) ? expf(score - mx) : 0.f;
  const float s1 = blk_sum256(e1, red, tid);
  const float pr = e1 / s1;
  // mask + softmax #2
  const float v2  = (tid < Nd && adjn[n * Nd + tid] != 0.f) ? pr : -1e30f;
  const float mx2 = blk_max256(v2, red, tid);
  const float e2  = (tid < Nd && v2 > -1e29f) ? expf(v2 - mx2) : 0.f;
  const float s2  = blk_sum256(e2, red, tid);
  if (tid < Nd) attn_out[((long)p * Nd + n) * Nd + tid] = e2 / s2;
}

// token-major v (B,T,N,C) -> reference flat layout (B,C,T,N)
__global__ void vperm_kernel(const float* __restrict__ vt, float* __restrict__ vr) {
  const long i = (long)blockIdx.x * blockDim.x + threadIdx.x;
  if (i >= (long)TOKd * Cd) return;
  const int  c   = (int)(i & 63);
  const long tok = i >> 6;
  const int  n   = (int)(tok % Nd);
  const long bt  = tok / Nd;
  const int  t   = (int)(bt % Td);
  const int  b   = (int)(bt / Td);
  vr[(((long)(b * Cd + c)) * Td + t) * Nd + n] = vt[i];
}

// dst = LayerNorm(a + addend) * w + bias   (per token over C=64)
// mode 0: addend token-major; mode 1: addend in sp layout [(bt*64+c)*207+n]
__global__ void add_ln_kernel(const float* __restrict__ a, const float* __restrict__ badd,
                              int mode, const float* __restrict__ w, const float* __restrict__ bias,
                              float* __restrict__ dst) {
  const long tok = blockIdx.x;
  const int  c   = threadIdx.x;  // 0..63
  __shared__ float red[64];
  float add;
  if (mode == 0) add = badd[tok * Cd + c];
  else {
    const int  n  = (int)(tok % Nd);
    const long bt = tok / Nd;
    add = badd[(bt * Cd + c) * Nd + n];
  }
  const float v = a[tok * Cd + c] + add;
  red[c] = v; __syncthreads();
  for (int o = 32; o > 0; o >>= 1) { if (c < o) red[c] += red[c + o]; __syncthreads(); }
  const float mean = red[0] * (1.f / 64.f); __syncthreads();
  const float d = v - mean;
  red[c] = d * d; __syncthreads();
  for (int o = 32; o > 0; o >>= 1) { if (c < o) red[c] += red[c + o]; __syncthreads(); }
  const float var = red[0] * (1.f / 64.f); __syncthreads();
  dst[tok * Cd + c] = d * rsqrtf(var + 1e-5f) * w[c] + bias[c];
}

__global__ void add_kernel(const float* __restrict__ a, const float* __restrict__ b,
                           float* __restrict__ c, long n) {
  const long i = (long)blockIdx.x * blockDim.x + threadIdx.x;
  if (i < n) c[i] = a[i] + b[i];
}

// Temporal attention per (b, n, head): T=24, HD=16. tq/tk/tv token-major.
__global__ void temporal_attn_kernel(const float* __restrict__ tq, const float* __restrict__ tk,
                                     const float* __restrict__ tv, float* __restrict__ outh) {
  const int idx  = blockIdx.x;
  const int head = idx & 3;
  const int n    = (idx >> 2) % Nd;
  const int b    = idx / (NHd * Nd);
  const int t    = threadIdx.x;
  if (t >= Td) return;
  const long base = ((long)((b * Td + t) * Nd + n)) * Cd + head * HDd;
  float qv[HDd];
#pragma unroll
  for (int d = 0; d < HDd; ++d) qv[d] = tq[base + d];
  float sc[Td];
  float mx = -1e30f;
#pragma unroll
  for (int s = 0; s < Td; ++s) {
    const long kb = ((long)((b * Td + s) * Nd + n)) * Cd + head * HDd;
    float d = 0.f;
#pragma unroll
    for (int dd = 0; dd < HDd; ++dd) d += qv[dd] * tk[kb + dd];
    sc[s] = d * 0.25f;  // 1/sqrt(16)
    mx = fmaxf(mx, sc[s]);
  }
  float sum = 0.f;
#pragma unroll
  for (int s = 0; s < Td; ++s) { sc[s] = expf(sc[s] - mx); sum += sc[s]; }
  const float inv = 1.f / sum;
#pragma unroll
  for (int d = 0; d < HDd; ++d) {
    float o = 0.f;
#pragma unroll
    for (int s = 0; s < Td; ++s)
      o += sc[s] * tv[((long)((b * Td + s) * Nd + n)) * Cd + head * HDd + d];
    outh[base + d] = o * inv;
  }
}

// h (B,T,N,C) -> seq (T, B*N, C)
__global__ void seq_transpose_kernel(const float* __restrict__ h, float* __restrict__ seq) {
  const long i = (long)blockIdx.x * blockDim.x + threadIdx.x;
  if (i >= (long)TOKd * Cd) return;
  const int  c   = (int)(i & 63);
  const long tok = i >> 6;
  const int  n   = (int)(tok % Nd);
  const long bt  = tok / Nd;
  const int  t   = (int)(bt % Td);
  const int  b   = (int)(bt / Td);
  seq[((long)t * BNd + b * Nd + n) * Cd + c] = h[i];
}

__global__ void zero_kernel(float* __restrict__ p, long n) {
  const long i = (long)blockIdx.x * blockDim.x + threadIdx.x;
  if (i < n) p[i] = 0.f;
}

// GRU cell: r/z/n gates, update hstate and seq[t]
__global__ void gru_cell_kernel(const float* __restrict__ xp_t, const float* __restrict__ hg,
                                float* __restrict__ hstate, float* __restrict__ seq_t) {
  const long i = (long)blockIdx.x * blockDim.x + threadIdx.x;
  if (i >= (long)BNd * Cd) return;
  const int row = (int)(i >> 6), c = (int)(i & 63);
  const float xr = xp_t[(long)row * 192 + c];
  const float xz = xp_t[(long)row * 192 + 64 + c];
  const float xn = xp_t[(long)row * 192 + 128 + c];
  const float hr = hg[(long)row * 192 + c];
  const float hz = hg[(long)row * 192 + 64 + c];
  const float hn = hg[(long)row * 192 + 128 + c];
  const float hp = hstate[i];
  const float r = 1.f / (1.f + expf(-(xr + hr)));
  const float z = 1.f / (1.f + expf(-(xz + hz)));
  const float nn = tanhf(xn + r * hn);
  const float hnew = (1.f - z) * nn + z * hp;
  hstate[i] = hnew;
  seq_t[i]  = hnew;
}

// seq (T,BN,C) -> flat (BN, C*T) with flat[row, c*T + t]
__global__ void flat_transpose_kernel(const float* __restrict__ seq, float* __restrict__ flat) {
  const long i = (long)blockIdx.x * blockDim.x + threadIdx.x;
  if (i >= (long)TOKd * Cd) return;
  const int  c   = (int)(i & 63);
  const long r   = i >> 6;
  const int  row = (int)(r % BNd);
  const int  t   = (int)(r / BNd);
  flat[(long)row * (Cd * Td) + c * Td + t] = seq[i];
}

// ptmp (BN,12) -> out (B,HOR,N)
__global__ void out_reorder_kernel(const float* __restrict__ ptmp, float* __restrict__ out) {
  const int i = blockIdx.x * blockDim.x + threadIdx.x;
  if (i >= BNd * HORd) return;
  const int hor = i % HORd, row = i / HORd;
  const int n = row % Nd, b = row / Nd;
  out[((long)(b * HORd + hor)) * Nd + n] = ptmp[i];
}

// ---------------------------------------------------------------------------
extern "C" void kernel_launch(void* const* d_in, const int* in_sizes, int n_in,
                              void* d_out, int out_size, void* d_ws, size_t ws_size,
                              hipStream_t stream) {
  (void)in_sizes; (void)n_in; (void)out_size; (void)ws_size;
  const float* x      = (const float*)d_in[0];
  const float* adj    = (const float*)d_in[1];
  const float* nodeE  = (const float*)d_in[2];
  const float* hour_w = (const float*)d_in[3];
  const float* hour_b = (const float*)d_in[4];
  const float* day_w  = (const float*)d_in[5];
  const float* day_b  = (const float*)d_in[6];
  const float* wk_w   = (const float*)d_in[7];
  const float* wk_b   = (const float*)d_in[8];
  const float* in_w   = (const float*)d_in[9];
  const float* in_b   = (const float*)d_in[10];
  const float* sa_qw  = (const float*)d_in[11];
  const float* sa_qb  = (const float*)d_in[12];
  const float* sa_kw  = (const float*)d_in[13];
  const float* sa_kb  = (const float*)d_in[14];
  const float* sa_vw  = (const float*)d_in[15];
  const float* sa_vb  = (const float*)d_in[16];
  const float* ta_qw  = (const float*)d_in[17];
  const float* ta_qb  = (const float*)d_in[18];
  const float* ta_kw  = (const float*)d_in[19];
  const float* ta_kb  = (const float*)d_in[20];
  const float* ta_vw  = (const float*)d_in[21];
  const float* ta_vb  = (const float*)d_in[22];
  const float* ta_ow  = (const float*)d_in[23];
  const float* ta_ob  = (const float*)d_in[24];
  const float* ln1_w  = (const float*)d_in[25];
  const float* ln1_b  = (const float*)d_in[26];
  const float* ln2_w  = (const float*)d_in[27];
  const float* ln2_b  = (const float*)d_in[28];
  const float* ffn_w1 = (const float*)d_in[29];
  const float* ffn_b1 = (const float*)d_in[30];
  const float* ffn_w2 = (const float*)d_in[31];
  const float* ffn_b2 = (const float*)d_in[32];
  const float* gwih   = (const float*)d_in[33];
  const float* gwhh   = (const float*)d_in[34];
  const float* gbih   = (const float*)d_in[35];
  const float* gbhh   = (const float*)d_in[36];
  const float* pred_w = (const float*)d_in[37];
  const float* pred_b = (const float*)d_in[38];

  float* out = (float*)d_out;
  float* ws  = (float*)d_ws;
  const long HC = (long)TOKd * Cd;        // 5,087,232 floats
  float* h    = ws;                       // activations (B,T,N,C)
  float* hl   = ws + HC;                  // post-LN2 / GRU seq (T,BN,C)
  float* b2   = ws + 2 * HC;              // v_tmp / tq / ta_proj / flat
  float* b3   = ws + 3 * HC;              // sp / tk / ffn out
  float* big  = ws + 4 * HC;              // 4*HC: v_ref | tv + ta_heads | f1 | xp
  float* qbuf = ws + 8 * HC;              // TOK*8
  float* kbuf = qbuf + (long)TOKd * 8;    // TOK*8
  float* adjn = kbuf + (long)TOKd * 8;    // 207*207
  float* hst  = adjn + (long)Nd * Nd;     // BN*64
  float* hg   = hst + (long)BNd * Cd;     // BN*192
  float* ptmp = hg + (long)BNd * 192;     // BN*12

  const int EW = (int)((HC + 255) / 256); // elementwise grid over TOK*C

  adjnorm_kernel<<<Nd, 256, 0, stream>>>(adj, adjn);
  embed_kernel<<<TOKd, Cd, 0, stream>>>(x, nodeE, hour_w, hour_b, day_w, day_b,
                                        wk_w, wk_b, in_w, in_b, h);

  for (int l = 0; l < 3; ++l) {
    float* attn_l = out + (long)Bd * HORd * Nd + (long)l * BTd * Nd * Nd;
    // spatial projections (x @ W^T + b)
    gemm(h, sa_qw + (long)l * 8 * Cd,  sa_qb + l * 8,  qbuf, TOKd, 8,  Cd, 1, 0, 0, 0, false, stream);
    gemm(h, sa_kw + (long)l * 8 * Cd,  sa_kb + l * 8,  kbuf, TOKd, 8,  Cd, 1, 0, 0, 0, false, stream);
    gemm(h, sa_vw + (long)l * Cd * Cd, sa_vb + l * Cd, b2,   TOKd, Cd, Cd, 1, 0, 0, 0, false, stream);
    vperm_kernel<<<EW, 256, 0, stream>>>(b2, big);
    // scores + double softmax with adjacency mask -> attn matrices (in d_out)
    spatial_attn_kernel<<<dim3(BTd, Nd), 256, 0, stream>>>(qbuf, kbuf, adjn, attn_l);
    // sp_p = V_p (64x207) @ attn_p^T  : batched NT WMMA GEMM
    gemm(big, attn_l, nullptr, b3, Cd, Nd, Nd, BTd,
         (long)Cd * Nd, (long)Nd * Nd, (long)Cd * Nd, false, stream);
    // h = LN1(h + sp)
    add_ln_kernel<<<TOKd, Cd, 0, stream>>>(h, b3, 1, ln1_w + l * Cd, ln1_b + l * Cd, h);
    // temporal attention
    gemm(h, ta_qw + (long)l * Cd * Cd, ta_qb + l * Cd, b2,  TOKd, Cd, Cd, 1, 0, 0, 0, false, stream);
    gemm(h, ta_kw + (long)l * Cd * Cd, ta_kb + l * Cd, b3,  TOKd, Cd, Cd, 1, 0, 0, 0, false, stream);
    gemm(h, ta_vw + (long)l * Cd * Cd, ta_vb + l * Cd, big, TOKd, Cd, Cd, 1, 0, 0, 0, false, stream);
    temporal_attn_kernel<<<Bd * Nd * NHd, 32, 0, stream>>>(b2, b3, big, big + HC);
    gemm(big + HC, ta_ow + (long)l * Cd * Cd, ta_ob + l * Cd, b2, TOKd, Cd, Cd, 1, 0, 0, 0, false, stream);
    // hl = LN2(h + ta_proj)
    add_ln_kernel<<<TOKd, Cd, 0, stream>>>(h, b2, 0, ln2_w + l * Cd, ln2_b + l * Cd, hl);
    // FFN: relu(hl @ w1^T + b1) @ w2^T + b2 ; h = hl + ffn
    gemm(hl,  ffn_w1 + (long)l * 256 * Cd, ffn_b1 + l * 256, big, TOKd, 256, Cd,  1, 0, 0, 0, true,  stream);
    gemm(big, ffn_w2 + (long)l * Cd * 256, ffn_b2 + l * Cd,  b3,  TOKd, Cd,  256, 1, 0, 0, 0, false, stream);
    add_kernel<<<EW, 256, 0, stream>>>(hl, b3, h, HC);
  }

  // GRU (2 layers) over seq (T, B*N, C)
  seq_transpose_kernel<<<EW, 256, 0, stream>>>(h, hl);
  for (int gl = 0; gl < 2; ++gl) {
    gemm(hl, gwih + (long)gl * 192 * Cd, gbih + gl * 192, big, TOKd, 192, Cd, 1, 0, 0, 0, false, stream);
    zero_kernel<<<(BNd * Cd + 255) / 256, 256, 0, stream>>>(hst, (long)BNd * Cd);
    for (int t = 0; t < Td; ++t) {
      gemm(hst, gwhh + (long)gl * 192 * Cd, gbhh + gl * 192, hg, BNd, 192, Cd, 1, 0, 0, 0, false, stream);
      gru_cell_kernel<<<(BNd * Cd + 255) / 256, 256, 0, stream>>>(
          big + (long)t * BNd * 192, hg, hst, hl + (long)t * BNd * Cd);
    }
  }

  // prediction head
  flat_transpose_kernel<<<EW, 256, 0, stream>>>(hl, b2);
  gemm(b2, pred_w, pred_b, ptmp, BNd, HORd, Cd * Td, 1, 0, 0, 0, false, stream);
  out_reorder_kernel<<<(BNd * HORd + 255) / 256, 256, 0, stream>>>(ptmp, out);
}
